// RNNIOB_91036126806302
// MI455X (gfx1250) — compile-verified, matrix operand order checked by reading, not run on hardware
//
#include <hip/hip_runtime.h>
#include <hip/hip_bf16.h>

// ---------------------------------------------------------------------------
// Model dims
// ---------------------------------------------------------------------------
#define SEQ    256
#define BATCH  64
#define EMBD   256
#define HID    512
#define OUTD   18
#define SB     (SEQ * BATCH)   // 16384 rows

typedef __attribute__((ext_vector_type(16))) __bf16 v16bf;
typedef __attribute__((ext_vector_type(2)))  __bf16 v2bf;
typedef __attribute__((ext_vector_type(8)))  float  v8f;

// ---------------------------------------------------------------------------
// bf16 helpers — use native casts so the compiler emits v_cvt_pk_bf16_f32
// instead of the 4-VALU software RNE sequence.
// ---------------------------------------------------------------------------
__device__ __forceinline__ unsigned short f2bf(float a) {
  __bf16 h = (__bf16)a;
  return __builtin_bit_cast(unsigned short, h);
}
__device__ __forceinline__ unsigned int f2bf_pack(float a, float b) {
  v2bf p;
  p.x = (__bf16)a;
  p.y = (__bf16)b;
  return __builtin_bit_cast(unsigned int, p);
}
__device__ __forceinline__ float bf2f(unsigned short h) {
  return __uint_as_float(((unsigned int)h) << 16);
}

__device__ __forceinline__ v8f wmma_bf16(v16bf a, v16bf b, v8f c) {
  // D(f32 16x16) = A(16x32 bf16) * B(32x16 bf16) + C
  return __builtin_amdgcn_wmma_f32_16x16x32_bf16(false, a, false, b,
                                                 (short)0, c, false, false);
}

// CDNA5 16-bit A/B operand fragment from a row-major bf16 matrix.
// Lane layout (ISA 7.12.2): lanes 0-15 -> row=lane, K pairs {0..7,16..23};
// lanes 16-31 -> row=lane-16, K pairs {8..15,24..31}. Pairs are contiguous
// dwords -> j=0..3 and j=4..7 merge into b128 global loads.
__device__ __forceinline__ v16bf load_frag_bf16(const unsigned short* tile,
                                                int ld, int lane) {
  const int r  = lane & 15;
  const int kh = (lane >> 4) << 3;
  const unsigned short* row = tile + (long long)r * ld;
  union { v16bf v; unsigned int u[8]; } f;
#pragma unroll
  for (int j = 0; j < 8; ++j) {
    const int k = ((j >> 2) << 4) + kh + ((j & 3) << 1);
    f.u[j] = *(const unsigned int*)(row + k);
  }
  return f.v;
}

// ---------------------------------------------------------------------------
// Weight f32 -> bf16 conversion with optional row padding (for fc_w 18->32)
// ---------------------------------------------------------------------------
__global__ void rnn_cvt_bf16_pad(const float* __restrict__ src,
                                 unsigned short* __restrict__ dst,
                                 int rows, int cols, int rows_pad) {
  int i = blockIdx.x * blockDim.x + threadIdx.x;
  if (i >= rows_pad * cols) return;
  int r = i / cols;
  dst[i] = (r < rows) ? f2bf(src[i]) : (unsigned short)0;
}

// ---------------------------------------------------------------------------
// Fused embedding gather + emfc GEMM:  pe[m][n] = emb[text[m]] @ emfc_w^T + b
// One wave per 16x16 C tile, K=256 in bf16 WMMA steps of 32.
// ---------------------------------------------------------------------------
__global__ __launch_bounds__(256) void rnn_embed_emfc(
    const int* __restrict__ text, const float* __restrict__ table,
    const unsigned short* __restrict__ Wbf,   // emfc_w bf16 [256][256]
    const float* __restrict__ bias,           // emfc_b [256]
    unsigned short* __restrict__ pe)          // out bf16 [SB][256]
{
  const int lane = threadIdx.x & 31;
  const int wave = blockIdx.x * (blockDim.x >> 5) + (threadIdx.x >> 5);
  const int NT = EMBD / 16;
  const int mt = wave / NT, nt = wave % NT;
  if (mt * 16 >= SB) return;                     // wave-uniform (EXEC all-1s)
  const int m0 = mt * 16, n0 = nt * 16;
  const int r = lane & 15, kh = (lane >> 4) << 3;
  // gathered embedding row for this lane (padding_idx row is zero in table)
  const float* arow = table + (long long)text[m0 + r] * EMBD;

  v8f c = {};
#pragma unroll 2
  for (int k0 = 0; k0 < EMBD; k0 += 32) {
    union { v16bf v; unsigned int u[8]; } af;
#pragma unroll
    for (int j = 0; j < 8; ++j) {
      const int k = k0 + ((j >> 2) << 4) + kh + ((j & 3) << 1);
      af.u[j] = f2bf_pack(arow[k], arow[k + 1]);   // v_cvt_pk_bf16_f32
    }
    v16bf b = load_frag_bf16(Wbf + (long long)n0 * EMBD + k0, EMBD, lane);
    c = wmma_bf16(af.v, b, c);
  }
  const int  n  = n0 + (lane & 15);
  const int  mo = (lane >> 4) << 3;
  const float bn = bias[n];
#pragma unroll
  for (int rr = 0; rr < 8; ++rr)
    pe[(long long)(m0 + mo + rr) * EMBD + n] = f2bf(c[rr] + bn);
}

// ---------------------------------------------------------------------------
// Generic bf16 GEMM: out[m][n] = A[m][:] @ W[n][:] + bias1[n] (+ bias2[n])
// A may be a feature-concat of A1|A2 split at Ksplit (bidirectional concat).
// Output either bf16 (xp buffers) or f32 (final logits, n < Nvalid guard).
// ---------------------------------------------------------------------------
__global__ __launch_bounds__(256) void rnn_gemm_bf16(
    const unsigned short* __restrict__ A1, const unsigned short* __restrict__ A2,
    int ldA, int Ksplit,
    const unsigned short* __restrict__ W, int ldW,       // row-major [N][ldW]
    const float* __restrict__ bias1, const float* __restrict__ bias2,
    unsigned short* __restrict__ out_bf, float* __restrict__ out_f,
    int M, int N, int K, int Nvalid, int ldO)
{
  const int lane = threadIdx.x & 31;
  const int wave = blockIdx.x * (blockDim.x >> 5) + (threadIdx.x >> 5);
  const int NT = N >> 4;
  const int mt = wave / NT, nt = wave % NT;
  if (mt * 16 >= M) return;                       // wave-uniform
  const int m0 = mt * 16, n0 = nt * 16;

  v8f c = {};
#pragma unroll 2
  for (int k0 = 0; k0 < K; k0 += 32) {
    const unsigned short* abase =
        (k0 >= Ksplit) ? (A2 + (long long)m0 * ldA + (k0 - Ksplit))
                       : (A1 + (long long)m0 * ldA + k0);
    v16bf a = load_frag_bf16(abase, ldA, lane);
    v16bf b = load_frag_bf16(W + (long long)n0 * ldW + k0, ldW, lane);
    c = wmma_bf16(a, b, c);
  }
  const int n  = n0 + (lane & 15);
  const int mo = (lane >> 4) << 3;
  float bn = (n < Nvalid) ? bias1[n] : 0.0f;
  if (bias2 && n < Nvalid) bn += bias2[n];
#pragma unroll
  for (int rr = 0; rr < 8; ++rr) {
    const int m = m0 + mo + rr;
    const float v = c[rr] + bn;
    if (out_bf)           out_bf[(long long)m * ldO + n] = f2bf(v);
    else if (n < Nvalid)  out_f [(long long)m * ldO + n] = v;
  }
}

// ---------------------------------------------------------------------------
// Recurrent scan: one persistent workgroup per direction.
//   h_t = relu(xp_t + h_{t-1} @ Whh^T),  B=64, H=512, 256 sequential steps.
// h lives in LDS as bf16 with an XOR dword swizzle (row stride 256 dwords is
// a multiple of 64 banks -> swizzle avoids 16-way conflicts on A fragments).
// 32 waves: wave w owns N-stripe [16w,16w+16), computes 4 M-tiles -> the full
// 64x512 step is 128 WMMA tiles, K=512 => 2048 v_wmma per timestep per dir.
// Whh (bf16, 512KB/dir) streams from L2 each step.
// ---------------------------------------------------------------------------
__device__ __forceinline__ v16bf load_frag_h_lds(const unsigned int* hb,
                                                 int mbase, int k0, int lane) {
  const int r  = mbase + (lane & 15);
  const int kh = (lane >> 4) << 3;
  const int sw = r & 63;
  union { v16bf v; unsigned int u[8]; } f;
#pragma unroll
  for (int j = 0; j < 8; ++j) {
    const int k  = k0 + ((j >> 2) << 4) + kh + ((j & 3) << 1);
    const int kd = k >> 1;                      // dword index within row
    f.u[j] = hb[r * (HID / 2) + (kd ^ sw)];
  }
  return f.v;
}

__global__ __launch_bounds__(1024) void rnn_scan(
    const unsigned short* __restrict__ xp,     // bf16 [2][SEQ][BATCH][HID]
    const unsigned short* __restrict__ Whh,    // bf16 [2][HID][HID]
    unsigned short* __restrict__ hout)         // bf16 [2][SEQ][BATCH][HID]
{
  const int d    = blockIdx.x;                 // direction (0=fwd, 1=bwd)
  const int lane = threadIdx.x & 31;
  const int w    = threadIdx.x >> 5;           // 0..31 -> N stripe
  const int n0   = w * 16;

  __shared__ unsigned int hb[BATCH * HID / 2]; // 64KB bf16 hidden state
  for (int i = threadIdx.x; i < BATCH * HID / 2; i += blockDim.x) hb[i] = 0u;
  __syncthreads();

  const unsigned short* Wd  = Whh + (long long)d * HID * HID;
  const unsigned short* xpd = xp  + (long long)d * SEQ * BATCH * HID;
  unsigned short*       hd  = hout + (long long)d * SEQ * BATCH * HID;

  const int nlane = n0 + (lane & 15);
  const int mo    = (lane >> 4) << 3;
  unsigned short* hb_us = (unsigned short*)hb;

  for (int i = 0; i < SEQ; ++i) {
    const int t = d ? (SEQ - 1 - i) : i;
    const unsigned short* xpt = xpd + (long long)t * BATCH * HID;
    if (i + 1 < SEQ) {  // prefetch next timestep's xp (global_prefetch_b8)
      const int tn = d ? (SEQ - 2 - i) : (i + 1);
      __builtin_prefetch(xpd + (long long)tn * BATCH * HID + (threadIdx.x << 5), 0, 1);
    }

    v8f c0 = {}, c1 = {}, c2 = {}, c3 = {};
#pragma unroll 4
    for (int k0 = 0; k0 < HID; k0 += 32) {
      v16bf b  = load_frag_bf16(Wd + (long long)n0 * HID + k0, HID, lane);
      v16bf a0 = load_frag_h_lds(hb,  0, k0, lane);
      v16bf a1 = load_frag_h_lds(hb, 16, k0, lane);
      v16bf a2 = load_frag_h_lds(hb, 32, k0, lane);
      v16bf a3 = load_frag_h_lds(hb, 48, k0, lane);
      c0 = wmma_bf16(a0, b, c0);
      c1 = wmma_bf16(a1, b, c1);
      c2 = wmma_bf16(a2, b, c2);
      c3 = wmma_bf16(a3, b, c3);
    }
    __syncthreads();   // all reads of h_{t-1} done before overwriting

    unsigned short* hg = hd + (long long)t * BATCH * HID;
#pragma unroll
    for (int mt = 0; mt < 4; ++mt) {
      v8f& c = (mt == 0) ? c0 : (mt == 1) ? c1 : (mt == 2) ? c2 : c3;
#pragma unroll
      for (int rr = 0; rr < 8; ++rr) {
        const int m = mt * 16 + mo + rr;
        float v = c[rr] + bf2f(xpt[m * HID + nlane]);
        v = fmaxf(v, 0.0f);
        const unsigned short hv = f2bf(v);
        const int dw = m * (HID / 2) + ((nlane >> 1) ^ (m & 63));
        hb_us[dw * 2 + (nlane & 1)] = hv;      // swizzled LDS write-back
        hg[m * HID + nlane] = hv;              // layer output for next GEMM
      }
    }
    __syncthreads();   // h_t fully written before next step reads it
  }
}

// ---------------------------------------------------------------------------
// Host-side orchestration
// ---------------------------------------------------------------------------
static inline size_t alignup(size_t x) { return (x + 255) & ~(size_t)255; }

extern "C" void kernel_launch(void* const* d_in, const int* in_sizes, int n_in,
                              void* d_out, int out_size, void* d_ws, size_t ws_size,
                              hipStream_t stream) {
  (void)in_sizes; (void)n_in; (void)out_size; (void)ws_size;

  const int*   text    = (const int*)  d_in[0];
  const float* table   = (const float*)d_in[1];
  const float* emfc_w  = (const float*)d_in[2];
  const float* emfc_b  = (const float*)d_in[3];
  const float* w_ih0   = (const float*)d_in[4];
  const float* w_hh0   = (const float*)d_in[5];
  const float* b_ih0   = (const float*)d_in[6];
  const float* b_hh0   = (const float*)d_in[7];
  const float* w_ih1   = (const float*)d_in[8];
  const float* w_hh1   = (const float*)d_in[9];
  const float* b_ih1   = (const float*)d_in[10];
  const float* b_hh1   = (const float*)d_in[11];
  const float* fc_w    = (const float*)d_in[12];
  const float* fc_b    = (const float*)d_in[13];
  float* out = (float*)d_out;

  // Workspace carve (bf16 everywhere). Three reusable 32MB activation slots.
  char* w = (char*)d_ws;
  const size_t actBytes = (size_t)2 * SB * HID * sizeof(unsigned short); // 32MB
  unsigned short* slotA = (unsigned short*)w; w += alignup(actBytes);
  unsigned short* slotB = (unsigned short*)w; w += alignup(actBytes);
  unsigned short* slotC = (unsigned short*)w; w += alignup(actBytes);
  unsigned short* emfcw = (unsigned short*)w; w += alignup((size_t)EMBD*EMBD*2);
  unsigned short* wih0  = (unsigned short*)w; w += alignup((size_t)2*HID*EMBD*2);
  unsigned short* whh0  = (unsigned short*)w; w += alignup((size_t)2*HID*HID*2);
  unsigned short* wih1  = (unsigned short*)w; w += alignup((size_t)2*HID*2*HID*2);
  unsigned short* whh1  = (unsigned short*)w; w += alignup((size_t)2*HID*HID*2);
  unsigned short* fcw   = (unsigned short*)w; w += alignup((size_t)32*2*HID*2);

  unsigned short* pe  = slotA;   // bf16 [SB][256]
  unsigned short* xp0 = slotB;   // bf16 [2][SB][512]
  unsigned short* h0  = slotC;   // bf16 [2][SB][512]
  unsigned short* xp1 = slotA;   // reuse (pe dead after xp0)
  unsigned short* h1  = slotB;   // reuse (xp0 dead after scan0)

  // 1) weights -> bf16 (fc_w padded 18 -> 32 rows with zeros)
  auto cvt = [&](const float* s, unsigned short* dptr, int rows, int cols, int rp) {
    int tot = rp * cols;
    rnn_cvt_bf16_pad<<<(tot + 255) / 256, 256, 0, stream>>>(s, dptr, rows, cols, rp);
  };
  cvt(emfc_w, emfcw, EMBD,    EMBD,   EMBD);
  cvt(w_ih0,  wih0,  2*HID,   EMBD,   2*HID);
  cvt(w_hh0,  whh0,  2*HID,   HID,    2*HID);
  cvt(w_ih1,  wih1,  2*HID,   2*HID,  2*HID);
  cvt(w_hh1,  whh1,  2*HID,   HID,    2*HID);
  cvt(fc_w,   fcw,   OUTD,    2*HID,  32);

  // 2) embedding gather + emfc GEMM -> pe  (waves = 1024*16, 8 waves/block)
  rnn_embed_emfc<<<(SB/16) * (EMBD/16) / 8, 256, 0, stream>>>(
      text, table, emfcw, emfc_b, pe);

  // 3) layer-0 input projection: xp0[d] = pe @ w_ih0[d]^T + (b_ih0+b_hh0)[d]
  for (int d = 0; d < 2; ++d) {
    rnn_gemm_bf16<<<(SB/16) * (HID/16) / 8, 256, 0, stream>>>(
        pe, pe, EMBD, /*Ksplit=*/EMBD,
        wih0 + (size_t)d * HID * EMBD, EMBD,
        b_ih0 + d * HID, b_hh0 + d * HID,
        xp0 + (size_t)d * SB * HID, nullptr,
        SB, HID, EMBD, HID, HID);
  }

  // 4) layer-0 recurrent scan (one WGP per direction)
  rnn_scan<<<2, 1024, 0, stream>>>(xp0, whh0, h0);

  // 5) layer-1 input projection on concat(h0_fwd | h0_bwd), K = 1024
  for (int d = 0; d < 2; ++d) {
    rnn_gemm_bf16<<<(SB/16) * (HID/16) / 8, 256, 0, stream>>>(
        h0, h0 + (size_t)SB * HID, HID, /*Ksplit=*/HID,
        wih1 + (size_t)d * HID * 2 * HID, 2 * HID,
        b_ih1 + d * HID, b_hh1 + d * HID,
        xp1 + (size_t)d * SB * HID, nullptr,
        SB, HID, 2 * HID, HID, HID);
  }

  // 6) layer-1 recurrent scan
  rnn_scan<<<2, 1024, 0, stream>>>(xp1, whh1, h1);

  // 7) final FC: out = concat(h1) @ fc_w^T + fc_b  (N padded to 32, store n<18)
  rnn_gemm_bf16<<<(SB/16) * (32/16) / 8, 256, 0, stream>>>(
      h1, h1 + (size_t)SB * HID, HID, /*Ksplit=*/HID,
      fcw, 2 * HID,
      fc_b, nullptr,
      nullptr, out,
      SB, 32, 2 * HID, OUTD, OUTD);
}
